// HiLo_59657095741857
// MI455X (gfx1250) — matrix operand order repaired and checked by reading, not compiled, not run on hardware
//
#include <hip/hip_runtime.h>
#include <hip/hip_bf16.h>

typedef __attribute__((ext_vector_type(16))) _Float16 v16h;
typedef __attribute__((ext_vector_type(8)))  _Float16 v8h;
typedef __attribute__((ext_vector_type(8)))  float    v8f;
typedef __attribute__((ext_vector_type(4)))  float    v4f;

__device__ inline v8f vzero8() {
  v8f z = {0.f, 0.f, 0.f, 0.f, 0.f, 0.f, 0.f, 0.f};
  return z;
}

// ---------- WMMA fragment helpers (layouts per CDNA5 ISA 7.12.2) ----
// All callers guarantee 16B alignment of every fragment row -> b128 loads.

// A fragment: 16x32 (MxK) f16, source row-major a[m*lda + k].
// lanes 0-15: M=lane, elems 0..7 = K 0..7,  elems 8..15 = K 16..23
// lanes16-31: M=lane-16, elems 0..7 = K 8..15, elems 8..15 = K 24..31
__device__ inline v16h load_frag_a(const _Float16* a, int lda) {
  const int lane = threadIdx.x & 31;
  const int m = lane & 15;
  const int kb = (lane & 16) ? 8 : 0;
  const _Float16* row = a + (size_t)m * lda + kb;
  v8h lo = *(const v8h*)(row);
  v8h hi = *(const v8h*)(row + 16);
  v16h f;
#pragma unroll
  for (int e = 0; e < 8; ++e) {
    f[e] = lo[e];
    f[8 + e] = hi[e];
  }
  return f;
}

// B fragment: 32x16 (KxN) f16, source stored N-major b[n*ldb + k].
// lanes 0-15: N=lane, elems 0..7 = K 0..7, elems 8..15 = K 16..23 ; lanes 16-31: N=lane-16, K 8..15, 24..31
__device__ inline v16h load_frag_b_nk(const _Float16* b, int ldb) {
  const int lane = threadIdx.x & 31;
  const int n = lane & 15;
  const int kb = (lane & 16) ? 8 : 0;
  const _Float16* row = b + (size_t)n * ldb + kb;
  v8h lo = *(const v8h*)(row);
  v8h hi = *(const v8h*)(row + 16);
  v16h f;
#pragma unroll
  for (int e = 0; e < 8; ++e) {
    f[e] = lo[e];
    f[8 + e] = hi[e];
  }
  return f;
}

__device__ inline v8f wmma16(v16h a, v16h b, v8f c) {
  return __builtin_amdgcn_wmma_f32_16x16x32_f16(false, a, false, b, (short)0, c,
                                                false, false);
}

// ---------- Generic tiled WMMA GEMM: C[M,N] = A[M,K] @ W[N,K]^T (+bias) -------
// 128 threads / 4 waves; each wave owns a 16x64 slab of a 64x64 C tile.
template <bool A_F32, bool OUT_F16>
__global__ __launch_bounds__(128) void gemm_wmma(
    const void* __restrict__ Ap, int lda, const float* __restrict__ W, int ldw,
    const float* __restrict__ bias, void* __restrict__ Out, int ldo, int K) {
  __shared__ _Float16 lA[64 * 40] __attribute__((aligned(16)));
  __shared__ _Float16 lB[64 * 40] __attribute__((aligned(16)));
  const int tid = threadIdx.x;
  const int wave = tid >> 5;
  const int lane = tid & 31;
  const int row0 = blockIdx.x * 64;
  const int col0 = blockIdx.y * 64;
  const int r = tid >> 1;         // 0..63 tile row (A and W tiles)
  const int c0 = (tid & 1) * 16;  // 0 or 16

  v8f acc[4];
#pragma unroll
  for (int j = 0; j < 4; ++j) acc[j] = vzero8();

  for (int kt = 0; kt < K; kt += 32) {
    // ---- stage A tile 64x32 (f32->f16 or f16 copy), all-b128 traffic
    {
      v8h h0, h1;
      if (A_F32) {
        const float* src = (const float*)Ap + (size_t)(row0 + r) * lda + kt + c0;
        v4f f0 = *(const v4f*)(src + 0);
        v4f f1 = *(const v4f*)(src + 4);
        v4f f2 = *(const v4f*)(src + 8);
        v4f f3 = *(const v4f*)(src + 12);
#pragma unroll
        for (int e = 0; e < 4; ++e) {
          h0[e] = (_Float16)f0[e];
          h0[4 + e] = (_Float16)f1[e];
          h1[e] = (_Float16)f2[e];
          h1[4 + e] = (_Float16)f3[e];
        }
      } else {
        const _Float16* src =
            (const _Float16*)Ap + (size_t)(row0 + r) * lda + kt + c0;
        h0 = *(const v8h*)(src);
        h1 = *(const v8h*)(src + 8);
      }
      *(v8h*)&lA[r * 40 + c0] = h0;
      *(v8h*)&lA[r * 40 + c0 + 8] = h1;
    }
    // ---- stage W tile 64x32 (f32->f16)
    {
      const float* src = W + (size_t)(col0 + r) * ldw + kt + c0;
      v4f f0 = *(const v4f*)(src + 0);
      v4f f1 = *(const v4f*)(src + 4);
      v4f f2 = *(const v4f*)(src + 8);
      v4f f3 = *(const v4f*)(src + 12);
      v8h h0, h1;
#pragma unroll
      for (int e = 0; e < 4; ++e) {
        h0[e] = (_Float16)f0[e];
        h0[4 + e] = (_Float16)f1[e];
        h1[e] = (_Float16)f2[e];
        h1[4 + e] = (_Float16)f3[e];
      }
      *(v8h*)&lB[r * 40 + c0] = h0;
      *(v8h*)&lB[r * 40 + c0 + 8] = h1;
    }
    __syncthreads();
    v16h a = load_frag_a(&lA[wave * 16 * 40], 40);
#pragma unroll
    for (int j = 0; j < 4; ++j) {
      v16h b = load_frag_b_nk(&lB[j * 16 * 40], 40);
      acc[j] = wmma16(a, b, acc[j]);
    }
    __syncthreads();
  }

  const int nloc = lane & 15;
  const int mhi = (lane & 16) ? 8 : 0;
#pragma unroll
  for (int j = 0; j < 4; ++j) {
    int col = col0 + j * 16 + nloc;
    float bv = bias ? bias[col] : 0.f;
#pragma unroll
    for (int i = 0; i < 8; ++i) {
      int row = row0 + wave * 16 + mhi + i;
      float v = acc[j][i] + bv;
      if (OUT_F16)
        ((_Float16*)Out)[(size_t)row * ldo + col] = (_Float16)v;
      else
        ((float*)Out)[(size_t)row * ldo + col] = v;
    }
  }
}

// ---------- Hi-Fi windowed 2x2 attention (4-token softmax; VALU, ~0.4 GF) -----
// Yh: [32768][512] f16 (cols 0..255 = k heads x 64, 256..511 = v)
// Ah: [32768][256] f16 attention output (head-concat per token)
__global__ __launch_bounds__(256) void hifi_attn(const _Float16* __restrict__ Yh,
                                                 _Float16* __restrict__ Ah) {
  int tid = blockIdx.x * 256 + threadIdx.x;  // 32*256*4*4 = 131072
  int i = tid & 3;
  int h = (tid >> 2) & 3;
  int g = (tid >> 4) & 255;
  int b = tid >> 12;
  int hg = g >> 4, wg = g & 15;
  int t[4];
#pragma unroll
  for (int j = 0; j < 4; ++j)
    t[j] = (b * 32 + hg * 2 + (j >> 1)) * 32 + wg * 2 + (j & 1);

  const _Float16* kcol = Yh + h * 64;
  const _Float16* vcol = Yh + 256 + h * 64;

  v8h kiv[8];
#pragma unroll
  for (int c = 0; c < 8; ++c)
    kiv[c] = *(const v8h*)(kcol + (size_t)t[i] * 512 + c * 8);

  float s[4];
#pragma unroll
  for (int j = 0; j < 4; ++j) {
    const _Float16* kj = kcol + (size_t)t[j] * 512;
    float acc = 0.f;
#pragma unroll
    for (int c = 0; c < 8; ++c) {
      v8h kv = *(const v8h*)(kj + c * 8);
#pragma unroll
      for (int e = 0; e < 8; ++e) acc += (float)kiv[c][e] * (float)kv[e];
    }
    s[j] = acc * 0.125f;
  }
  float mx = fmaxf(fmaxf(s[0], s[1]), fmaxf(s[2], s[3]));
  float p[4], sum = 0.f;
#pragma unroll
  for (int j = 0; j < 4; ++j) {
    p[j] = __expf(s[j] - mx);
    sum += p[j];
  }
  float inv = 1.f / sum;
#pragma unroll
  for (int j = 0; j < 4; ++j) p[j] *= inv;

  _Float16* orow = Ah + (size_t)t[i] * 256 + h * 64;
#pragma unroll
  for (int c = 0; c < 8; ++c) {
    v8h vj[4];
#pragma unroll
    for (int j = 0; j < 4; ++j)
      vj[j] = *(const v8h*)(vcol + (size_t)t[j] * 512 + c * 8);
    v8h ov;
#pragma unroll
    for (int e = 0; e < 8; ++e) {
      float o = 0.f;
#pragma unroll
      for (int j = 0; j < 4; ++j) o += p[j] * (float)vj[j][e];
      ov[e] = (_Float16)o;
    }
    *(v8h*)(orow + c * 8) = ov;
  }
}

// ---------- Lo-Fi flash attention: per (b,h), 1024x1024 k-k attention ---------
// grid (8 qtiles-of-128, 4 heads, 32 batch); 256 thr = 8 waves, 16 q-rows/wave.
// 64-key chunks: 16 WMMA per chunk per wave; only the row-max needs a per-chunk
// cross-lane reduce (row-sum is kept as per-lane partials, reduced once at end).
// V is staged TRANSPOSED in LDS so P@V B-fragments are contiguous b128 loads.
__global__ __launch_bounds__(256) void lofi_attn(const _Float16* __restrict__ Yl,
                                                 _Float16* __restrict__ Al) {
  __shared__ _Float16 lK[64 * 72] __attribute__((aligned(16)));   // [key][d]
  __shared__ _Float16 lVt[64 * 72] __attribute__((aligned(16)));  // [d][key]
  __shared__ _Float16 lP[8][16 * 72] __attribute__((aligned(16)));

  const int tid = threadIdx.x;
  const int wave = tid >> 5;
  const int lane = tid & 31;
  const int h = blockIdx.y;
  const int b = blockIdx.z;
  const int q0 = blockIdx.x * 128 + wave * 16;

  const _Float16* Kbase = Yl + (size_t)b * 1024 * 512 + h * 64;  // [n][d] ld=512
  const _Float16* Vbase = Kbase + 256;

  // Queries are the K rows themselves (k-k attention): two 16x32 A fragments.
  v16h qa0 = load_frag_a(Kbase + (size_t)q0 * 512, 512);
  v16h qa1 = load_frag_a(Kbase + (size_t)q0 * 512 + 32, 512);

  v8f o[4];
#pragma unroll
  for (int t = 0; t < 4; ++t) o[t] = vzero8();
  float rmax[8], rsum[8];  // rsum = per-lane partial, reduced after the loop
#pragma unroll
  for (int i = 0; i < 8; ++i) {
    rmax[i] = -1e30f;
    rsum[i] = 0.f;
  }

  const int sr = tid >> 2;        // staging key row 0..63 (coalesced globals)
  const int sc = (tid & 3) * 16;  // staging dim col 0,16,32,48

  for (int j0 = 0; j0 < 1024; j0 += 64) {
    // cooperative stage of 64-key K/V chunk (two b128 global loads per array
    // per thread); V goes in transposed for contiguous P@V fragments
    {
      const _Float16* kp = Kbase + (size_t)(j0 + sr) * 512 + sc;
      const _Float16* vp = Vbase + (size_t)(j0 + sr) * 512 + sc;
      v8h k0 = *(const v8h*)(kp);
      v8h k1 = *(const v8h*)(kp + 8);
      v8h v0 = *(const v8h*)(vp);
      v8h v1 = *(const v8h*)(vp + 8);
      *(v8h*)&lK[sr * 72 + sc] = k0;
      *(v8h*)&lK[sr * 72 + sc + 8] = k1;
#pragma unroll
      for (int e = 0; e < 8; ++e) {
        lVt[(sc + e) * 72 + sr] = v0[e];
        lVt[(sc + 8 + e) * 72 + sr] = v1[e];
      }
    }
    __syncthreads();

    // S(16x64) = Q(16x64) @ Kchunk^T : 8 WMMA (4 key-subtiles x 2 d-halves)
    v8f s[4];
#pragma unroll
    for (int t = 0; t < 4; ++t) {
      s[t] = vzero8();
      s[t] = wmma16(qa0, load_frag_b_nk(&lK[t * 16 * 72], 72), s[t]);
      s[t] = wmma16(qa1, load_frag_b_nk(&lK[t * 16 * 72 + 32], 72), s[t]);
    }

    // Online softmax. D layout: elem i -> row (i + 8*(lane>=16)), col lane&15.
    // A row lives in one 16-lane half => xor-reduce max with masks 1,2,4,8.
#pragma unroll
    for (int i = 0; i < 8; ++i) {
      float a0 = s[0][i] * 0.125f, a1 = s[1][i] * 0.125f;
      float a2 = s[2][i] * 0.125f, a3 = s[3][i] * 0.125f;
      float mx = fmaxf(fmaxf(a0, a1), fmaxf(a2, a3));
#pragma unroll
      for (int off = 1; off < 16; off <<= 1)
        mx = fmaxf(mx, __shfl_xor(mx, off, 32));
      float mnew = fmaxf(rmax[i], mx);
      float corr = __expf(rmax[i] - mnew);
      rmax[i] = mnew;
      float p0 = __expf(a0 - mnew);
      float p1 = __expf(a1 - mnew);
      float p2 = __expf(a2 - mnew);
      float p3 = __expf(a3 - mnew);
      rsum[i] = rsum[i] * corr + ((p0 + p1) + (p2 + p3));
#pragma unroll
      for (int t = 0; t < 4; ++t) o[t][i] *= corr;
      // stage P (D layout) to per-wave LDS for relayout into A fragments
      int m = i + ((lane & 16) ? 8 : 0);
      _Float16* pr = &lP[wave][m * 72 + (lane & 15)];
      pr[0] = (_Float16)p0;
      pr[16] = (_Float16)p1;
      pr[32] = (_Float16)p2;
      pr[48] = (_Float16)p3;
    }
    // wave-local LDS RAW: lanes are lockstep, just drain the DS counter
    asm volatile("s_wait_dscnt 0" ::: "memory");
    v16h pa0 = load_frag_a(&lP[wave][0], 72);   // keys 0..31
    v16h pa1 = load_frag_a(&lP[wave][32], 72);  // keys 32..63
#pragma unroll
    for (int t = 0; t < 4; ++t) {
      o[t] = wmma16(pa0, load_frag_b_nk(&lVt[t * 16 * 72], 72), o[t]);
      o[t] = wmma16(pa1, load_frag_b_nk(&lVt[t * 16 * 72 + 32], 72), o[t]);
    }
    __syncthreads();
  }

  // deferred row-sum reduction (once, instead of per chunk)
#pragma unroll
  for (int i = 0; i < 8; ++i) {
    float v = rsum[i];
#pragma unroll
    for (int off = 1; off < 16; off <<= 1) v += __shfl_xor(v, off, 32);
    rsum[i] = 1.f / v;
  }
  _Float16* orow = Al + (size_t)(b * 1024 + q0) * 256 + h * 64;
  const int nloc = lane & 15;
  const int mhi = (lane & 16) ? 8 : 0;
#pragma unroll
  for (int t = 0; t < 4; ++t)
#pragma unroll
    for (int i = 0; i < 8; ++i)
      orow[(size_t)(mhi + i) * 256 + t * 16 + nloc] =
          (_Float16)(o[t][i] * rsum[i]);
}

// ------------------------------- launch ---------------------------------------
extern "C" void kernel_launch(void* const* d_in, const int* in_sizes, int n_in,
                              void* d_out, int out_size, void* d_ws, size_t ws_size,
                              hipStream_t stream) {
  const float* x       = (const float*)d_in[0];
  const float* Wh_qkv  = (const float*)d_in[1];
  const float* Wh_proj = (const float*)d_in[2];
  const float* bh_proj = (const float*)d_in[3];
  const float* Wl_kv   = (const float*)d_in[4];
  const float* Wl_proj = (const float*)d_in[5];
  const float* bl_proj = (const float*)d_in[6];
  // d_in[7]=H, d_in[8]=W (fixed 32)

  const int M = 32 * 1024;  // 32768 tokens
  _Float16* Yh = (_Float16*)d_ws;               // 32768 x 512 f16 (hifi k|v)
  _Float16* Yl = Yh + (size_t)M * 512;          // 32768 x 512 f16 (lofi k|v)
  _Float16* Ah = Yl + (size_t)M * 512;          // 32768 x 256 f16
  _Float16* Al = Ah + (size_t)M * 256;          // 32768 x 256 f16
  float* out = (float*)d_out;

  dim3 blk(128);
  // 1) hifi k,v: x @ Wh_qkv[256:768]^T  (q rows are dead code in k-k attention)
  gemm_wmma<true, true><<<dim3(M / 64, 8), blk, 0, stream>>>(
      x, 512, Wh_qkv + 256 * 512, 512, nullptr, Yh, 512, 512);
  // 2) lofi k,v: x @ Wl_kv^T
  gemm_wmma<true, true><<<dim3(M / 64, 8), blk, 0, stream>>>(
      x, 512, Wl_kv, 512, nullptr, Yl, 512, 512);
  // 3) hifi windowed attention
  hifi_attn<<<dim3(131072 / 256), dim3(256), 0, stream>>>(Yh, Ah);
  // 4) lofi flash attention
  lofi_attn<<<dim3(8, 4, 32), dim3(256), 0, stream>>>(Yl, Al);
  // 5) hi projection -> out[:, 0:256]
  gemm_wmma<false, false><<<dim3(M / 64, 4), blk, 0, stream>>>(
      Ah, 256, Wh_proj, 256, bh_proj, out, 512, 256);
  // 6) lo projection -> out[:, 256:512]
  gemm_wmma<false, false><<<dim3(M / 64, 4), blk, 0, stream>>>(
      Al, 256, Wl_proj, 256, bl_proj, out + 256, 512, 256);
}
